// GoluImageNet_39900246180333
// MI455X (gfx1250) — matrix operand order repaired and checked by reading, not compiled, not run on hardware
//
#include <hip/hip_runtime.h>
#include <math.h>

// ---------------------------------------------------------------------------
// GoluImageNet on MI455X (gfx1250): all matmuls via v_wmma_f32_16x16x32_bf16.
// f32 weights/activations -> bf16 WMMA A/B operands, f32 accumulate.
// Branch-free vectorized LDS staging (b128 global loads / b128 ds stores).
// ---------------------------------------------------------------------------

typedef __attribute__((ext_vector_type(16))) __bf16 v16bf;
typedef __attribute__((ext_vector_type(8)))  float  v8f;
typedef unsigned short bf16_t;

#define E_DIM    384
#define NHEAD    6
#define HDIM     64
#define TOK      196
#define ROWS_IMG 198
#define BATCH    16
#define GRIDW    14
#define LAYERS   6

#define TM 128
#define TN 64
#define TK 32
#define LDSS 40   // LDS row stride (ushorts): 80B rows, 16B-aligned b128 stores

__device__ __forceinline__ bf16_t f2bf(float f) {
  unsigned u = __float_as_uint(f);
  u += 0x7fffu + ((u >> 16) & 1u);          // round-to-nearest-even
  return (bf16_t)(u >> 16);
}
__device__ __forceinline__ float gelu_f(float x) {
  return 0.5f * x * (1.0f + erff(x * 0.70710678118654752f));
}

// ---------------------------------------------------------------------------
// Generic WMMA GEMM:  C[m,n] = act( A[m,:] * W[:,n] + bias[n] )  (+= optional)
//   A: bf16 [M x K] (K multiple of 32, rows 16B-aligned), row-mapped:
//        grow = (m/arpb)*abstr + abase + m%arpb
//   W: f32 row-major [K x ldw] (wtr=0, column wco+n) or [N x ldw] (wtr=1)
//   C: f32 (Cf) and/or bf16 (Cb), row mapping crpb/cbstr/cbase
// Tile 128x64, BK=32, 256 threads = 8 waves; wave w owns rows [16w,16w+16).
// ---------------------------------------------------------------------------
__global__ __launch_bounds__(256) void gemm_wmma_bf16(
    const bf16_t* __restrict__ A, const float* __restrict__ W,
    const float* __restrict__ bias,
    float* __restrict__ Cf, bf16_t* __restrict__ Cb,
    int M, int N, int K, int lda, int ldw, int ldc,
    int arpb, int abstr, int abase,
    int crpb, int cbstr, int cbase,
    int wco, int wtr, int act, int accum)
{
  __shared__ __align__(16) bf16_t As[TM][LDSS];   // [m][k]
  __shared__ __align__(16) bf16_t Bs[TN][LDSS];   // [n][k] (K contiguous)

  const int tid  = threadIdx.x;
  const int wave = tid >> 5;
  const int lane = tid & 31;
  const int m0 = blockIdx.y * TM;
  const int n0 = blockIdx.x * TN;
  const bool n_full = (n0 + TN <= N);

  const v8f z8 = {0.f, 0.f, 0.f, 0.f, 0.f, 0.f, 0.f, 0.f};
  v8f acc[4];
  acc[0] = z8; acc[1] = z8; acc[2] = z8; acc[3] = z8;

  // --- A staging map: thread -> (row, 16-elem half), branch-free mask ---
  const int ar    = tid >> 1;              // 0..127
  const int ahalf = (tid & 1) << 4;        // 0 or 16
  int am  = m0 + ar;
  int amc = min(am, M - 1);
  const size_t arow = (size_t)((amc / arpb) * abstr + abase + (amc % arpb)) * lda;
  const unsigned amask = (am < M) ? 0xFFFFFFFFu : 0u;

  // --- B staging maps ---
  const int bkc = tid >> 3;                // wtr=0: k row 0..31
  const int bn8 = (tid & 7) << 3;          // wtr=0: n offset 0..56
  const int tn  = tid >> 2;                // wtr=1: n 0..63
  const int tk8 = (tid & 3) << 3;          // wtr=1: k offset 0..24

  for (int kk = 0; kk < K; kk += TK) {
    // ---- stage A tile: 2x b128 loads, mask, 2x b128 LDS stores ----
    {
      const uint4* asrc = (const uint4*)(A + arow + kk + ahalf);
      uint4 a0 = asrc[0];
      uint4 a1 = asrc[1];
      a0.x &= amask; a0.y &= amask; a0.z &= amask; a0.w &= amask;
      a1.x &= amask; a1.y &= amask; a1.z &= amask; a1.w &= amask;
      *(uint4*)&As[ar][ahalf]     = a0;
      *(uint4*)&As[ar][ahalf + 8] = a1;
    }
    // ---- stage W tile -> bf16 ----
    if (!wtr) {
      if (n_full) {
        const float* wrow = W + (size_t)(kk + bkc) * ldw + wco + n0 + bn8;
        float4 f0 = *(const float4*)wrow;
        float4 f1 = *(const float4*)(wrow + 4);
        bf16_t h[8] = {f2bf(f0.x), f2bf(f0.y), f2bf(f0.z), f2bf(f0.w),
                       f2bf(f1.x), f2bf(f1.y), f2bf(f1.z), f2bf(f1.w)};
        #pragma unroll
        for (int j = 0; j < 8; ++j) Bs[bn8 + j][bkc] = h[j];
      } else {
        // N-tail tile (N=1000, N=4, ...): clamp + select, no OOB reads
        #pragma unroll
        for (int j = 0; j < 8; ++j) {
          int gn  = n0 + bn8 + j;
          int gnc = min(gn, N - 1);
          float wv = W[(size_t)(kk + bkc) * ldw + wco + gnc];
          Bs[bn8 + j][bkc] = (gn < N) ? f2bf(wv) : (bf16_t)0;
        }
      }
    } else {
      // transposed weight (patch conv): vector along K, packed b128 store
      int gn  = n0 + tn;
      int gnc = min(gn, N - 1);
      const float* wrow = W + (size_t)(wco + gnc) * ldw + kk + tk8;
      float4 f0 = *(const float4*)wrow;
      float4 f1 = *(const float4*)(wrow + 4);
      unsigned bm = (gn < N) ? 0xFFFFFFFFu : 0u;
      uint4 p;
      p.x = (((unsigned)f2bf(f0.x)) | ((unsigned)f2bf(f0.y) << 16)) & bm;
      p.y = (((unsigned)f2bf(f0.z)) | ((unsigned)f2bf(f0.w) << 16)) & bm;
      p.z = (((unsigned)f2bf(f1.x)) | ((unsigned)f2bf(f1.y) << 16)) & bm;
      p.w = (((unsigned)f2bf(f1.z)) | ((unsigned)f2bf(f1.w) << 16)) & bm;
      *(uint4*)&Bs[tn][tk8] = p;
    }
    if (kk + TK < K && !wtr)
      __builtin_prefetch(&W[(size_t)(kk + TK) * ldw + wco + n0], 0, 0);
    __syncthreads();

    // ---- fragments + WMMA (ISA 7.12.2 layouts) ----
    // A: lane l&15 = row, l>>4 selects K-octet
    int mrow = (wave << 4) + (lane & 15);
    int kb = (lane >> 4) << 3;
    v16bf af;
    #pragma unroll
    for (int i = 0; i < 8; ++i)
      af[i] = __builtin_bit_cast(__bf16, As[mrow][kb + i]);
    #pragma unroll
    for (int i = 0; i < 8; ++i)
      af[8 + i] = __builtin_bit_cast(__bf16, As[mrow][16 + kb + i]);

    // B: lane l&15 = column, l>>4 selects K-octet
    int ncl = lane & 15;
    int kh = (lane >> 4) << 3;
    #pragma unroll
    for (int j = 0; j < 4; ++j) {
      v16bf bfr;
      #pragma unroll
      for (int i = 0; i < 16; ++i)
        bfr[i] = __builtin_bit_cast(__bf16, Bs[j * 16 + ncl][kh + (i & 7) + ((i >> 3) << 4)]);
      acc[j] = __builtin_amdgcn_wmma_f32_16x16x32_bf16(
          false, af, false, bfr, (short)0, acc[j], false, false);
    }
    __syncthreads();
  }

  // ---- epilogue: D lane l&15 = N, VGPR r -> M = (l>>4)*8 + r ----
  int ncl = lane & 15;
  int mrb = m0 + (wave << 4) + ((lane >> 4) << 3);
  #pragma unroll
  for (int j = 0; j < 4; ++j) {
    int n = n0 + j * 16 + ncl;
    if (n >= N) continue;
    float bv = bias ? bias[n] : 0.f;
    #pragma unroll
    for (int r = 0; r < 8; ++r) {
      int m = mrb + r;
      if (m >= M) continue;
      float v = acc[j][r] + bv;
      if (act == 1)      v = gelu_f(v);
      else if (act == 2) v = 1.f / (1.f + __expf(-v));
      size_t grow = (size_t)((m / crpb) * cbstr + cbase + (m % crpb));
      if (Cf) {
        float* p = Cf + grow * (size_t)ldc + n;
        *p = accum ? (*p + v) : v;
      }
      if (Cb) Cb[grow * (size_t)ldc + n] = f2bf(v);
    }
  }
}

// ---------------------------------------------------------------------------
// Patch im2col: x[B,3,224,224] -> A[B*196, 768] bf16; float4 in, uint2 out.
// ---------------------------------------------------------------------------
__global__ __launch_bounds__(192) void im2col_patch(
    const float* __restrict__ x, bf16_t* __restrict__ Ap)
{
  int row = blockIdx.x;                 // b*196 + t
  int b = row / TOK, t = row % TOK;
  int gi = t / GRIDW, gj = t % GRIDW;
  int k = threadIdx.x * 4;              // 192 threads * 4 = 768
  int c = k >> 8, rem = k & 255;
  int ph = rem >> 4, pw = rem & 15;
  const float* src = x + (((size_t)b * 3 + c) * 224 + gi * 16 + ph) * 224
                       + gj * 16 + pw;
  float4 f = *(const float4*)src;
  uint2 p;
  p.x = ((unsigned)f2bf(f.x)) | ((unsigned)f2bf(f.y) << 16);
  p.y = ((unsigned)f2bf(f.z)) | ((unsigned)f2bf(f.w) << 16);
  *(uint2*)&Ap[(size_t)row * 768 + k] = p;
}

__global__ void init_tokens(float* __restrict__ z, const float* __restrict__ cls,
                            const float* __restrict__ box)
{
  int b = blockIdx.x;
  for (int i = threadIdx.x; i < E_DIM; i += blockDim.x) {
    z[((size_t)b * ROWS_IMG + 0) * E_DIM + i] = cls[i];
    z[((size_t)b * ROWS_IMG + 1) * E_DIM + i] = box[i];
  }
}

// ---------------------------------------------------------------------------
// LayerNorm over E=384 -> bf16. split=1: per-image rows 0,1 -> out_sp
// ([B*2,E]), patch rows -> out_pat ([B*196,E]); split=0: all rows -> out_pat.
// ---------------------------------------------------------------------------
__global__ __launch_bounds__(128) void layernorm_k(
    const float* __restrict__ z, const float* __restrict__ g,
    const float* __restrict__ bta, bf16_t* __restrict__ out_sp,
    bf16_t* __restrict__ out_pat, int split)
{
  int row = blockIdx.x;
  int tid = threadIdx.x;
  const float* x = z + (size_t)row * E_DIM;
  __shared__ float red[128];

  float s = 0.f;
  for (int i = tid; i < E_DIM; i += 128) s += x[i];
  red[tid] = s; __syncthreads();
  for (int o = 64; o > 0; o >>= 1) { if (tid < o) red[tid] += red[tid + o]; __syncthreads(); }
  float mean = red[0] * (1.f / E_DIM);
  __syncthreads();

  float v = 0.f;
  for (int i = tid; i < E_DIM; i += 128) { float d = x[i] - mean; v += d * d; }
  red[tid] = v; __syncthreads();
  for (int o = 64; o > 0; o >>= 1) { if (tid < o) red[tid] += red[tid + o]; __syncthreads(); }
  float rstd = rsqrtf(red[0] * (1.f / E_DIM) + 1e-5f);

  bf16_t* dst = nullptr;
  if (split) {
    int bi = row / ROWS_IMG, j = row % ROWS_IMG;
    if (j < 2) { if (out_sp)  dst = out_sp  + ((size_t)(bi * 2 + j)) * E_DIM; }
    else       { if (out_pat) dst = out_pat + ((size_t)(bi * TOK + j - 2)) * E_DIM; }
  } else {
    dst = out_pat + (size_t)row * E_DIM;
  }
  if (dst)
    for (int i = tid; i < E_DIM; i += 128)
      dst[i] = f2bf((x[i] - mean) * rstd * g[i] + bta[i]);
}

// ---------------------------------------------------------------------------
// Local 5x5-neighborhood attention with RoPE. One wave per (b,t,head);
// lanes own head-dims {lane, lane+32}. qkv: [B*196, 1152] f32.
// ---------------------------------------------------------------------------
__global__ __launch_bounds__(192) void local_attn(
    const float* __restrict__ qkv, bf16_t* __restrict__ aout)
{
  int bt = blockIdx.x;
  int b = bt / TOK, t = bt % TOK;
  int h = threadIdx.y;
  int lane = threadIdx.x;
  int gi = t / GRIDW, gj = t % GRIDW;

  int d = lane;                                   // 0..31
  float invf = __powf(10000.f, -(float)d * (1.f / 32.f));
  float sq, cq; __sincosf((float)t * invf, &sq, &cq);

  size_t rowq = (size_t)bt * 1152 + h * HDIM;
  float q1 = qkv[rowq + d];
  float q2 = qkv[rowq + d + 32];
  float qr1 = q1 * cq - q2 * sq;
  float qr2 = q1 * sq + q2 * cq;

  float w[25];
  #pragma unroll 1
  for (int n = 0; n < 25; ++n) {
    int dr = n / 5 - 2, dc = n % 5 - 2;
    int rr = min(max(gi + dr, 0), GRIDW - 1);
    int cc = min(max(gj + dc, 0), GRIDW - 1);
    int tnn = rr * GRIDW + cc;
    size_t rowk = (size_t)(b * TOK + tnn) * 1152 + 384 + h * HDIM;
    float sk, ck; __sincosf((float)tnn * invf, &sk, &ck);
    float k1 = qkv[rowk + d];
    float k2 = qkv[rowk + d + 32];
    float kr1 = k1 * ck - k2 * sk;
    float kr2 = k1 * sk + k2 * ck;
    float part = qr1 * kr1 + qr2 * kr2;
    for (int o = 16; o > 0; o >>= 1) part += __shfl_xor(part, o, 32);
    w[n] = part * 0.125f;                         // 1/sqrt(64)
  }

  float mx = -1e30f;
  #pragma unroll
  for (int n = 0; n < 25; ++n) mx = fmaxf(mx, w[n]);
  float sum = 0.f;
  #pragma unroll
  for (int n = 0; n < 25; ++n) { w[n] = __expf(w[n] - mx); sum += w[n]; }
  float inv = 1.f / sum;

  float a1 = 0.f, a2 = 0.f;
  #pragma unroll 1
  for (int n = 0; n < 25; ++n) {
    int dr = n / 5 - 2, dc = n % 5 - 2;
    int rr = min(max(gi + dr, 0), GRIDW - 1);
    int cc = min(max(gj + dc, 0), GRIDW - 1);
    int tnn = rr * GRIDW + cc;
    size_t rowv = (size_t)(b * TOK + tnn) * 1152 + 768 + h * HDIM;
    float p = w[n] * inv;
    a1 += p * qkv[rowv + d];
    a2 += p * qkv[rowv + d + 32];
  }
  size_t orow = (size_t)bt * E_DIM + h * HDIM;
  aout[orow + d] = f2bf(a1);
  aout[orow + d + 32] = f2bf(a2);
}

// ---------------------------------------------------------------------------
// Token attention: 2 special queries attend over 196 tokens per (b, head).
// tq: [B*2,384] f32 (not roped). tkv: [B*196,768] f32 (k roped, v raw).
// ---------------------------------------------------------------------------
__global__ __launch_bounds__(256) void token_attn(
    const float* __restrict__ tq, const float* __restrict__ tkv,
    bf16_t* __restrict__ so_b)
{
  int blk = blockIdx.x;                 // b*6 + h
  int b = blk / NHEAD, h = blk % NHEAD;
  int tid = threadIdx.x;
  __shared__ float sc[256];
  __shared__ float red[256];

  for (int qi = 0; qi < 2; ++qi) {
    float score = -1e30f;
    if (tid < TOK) {
      const float* tqp = tq + ((size_t)(b * 2 + qi)) * E_DIM + h * HDIM;
      size_t krow = (size_t)(b * TOK + tid) * 768 + h * HDIM;
      float acc = 0.f;
      for (int dd = 0; dd < 32; ++dd) {
        float invf = __powf(10000.f, -(float)dd * (1.f / 32.f));
        float s, c; __sincosf((float)tid * invf, &s, &c);
        float k1 = tkv[krow + dd], k2 = tkv[krow + dd + 32];
        acc += tqp[dd] * (k1 * c - k2 * s) + tqp[dd + 32] * (k1 * s + k2 * c);
      }
      score = acc * 0.125f;
    }
    sc[tid] = score; red[tid] = score; __syncthreads();
    for (int s = 128; s > 0; s >>= 1) { if (tid < s) red[tid] = fmaxf(red[tid], red[tid + s]); __syncthreads(); }
    float mx = red[0]; __syncthreads();
    float e = (tid < TOK) ? __expf(score - mx) : 0.f;
    sc[tid] = e; red[tid] = e; __syncthreads();
    for (int s = 128; s > 0; s >>= 1) { if (tid < s) red[tid] += red[tid + s]; __syncthreads(); }
    float denom = red[0]; __syncthreads();

    if (tid < HDIM) {
      float acc = 0.f;
      for (int t2 = 0; t2 < TOK; ++t2)
        acc += sc[t2] * tkv[(size_t)(b * TOK + t2) * 768 + 384 + h * HDIM + tid];
      so_b[((size_t)(b * 2 + qi)) * E_DIM + h * HDIM + tid] = f2bf(acc / denom);
    }
    __syncthreads();
  }
}

// z patch rows += po
__global__ void add_po(float* __restrict__ z, const float* __restrict__ po)
{
  int row = blockIdx.x;                 // b*196 + t
  int b = row / TOK, t = row % TOK;
  float* zp = z + ((size_t)(b * ROWS_IMG + 2 + t)) * E_DIM;
  const float* pp = po + (size_t)row * E_DIM;
  for (int i = threadIdx.x; i < E_DIM; i += blockDim.x) zp[i] += pp[i];
}

// ---------------------------------------------------------------------------
// Host-side orchestration
// ---------------------------------------------------------------------------
static inline void gemm(hipStream_t s, const bf16_t* A, const float* W,
                        const float* bias, float* Cf, bf16_t* Cb,
                        int M, int N, int K, int lda, int ldw, int ldc,
                        int arpb, int abstr, int abase,
                        int crpb, int cbstr, int cbase,
                        int wco, int wtr, int act, int accum)
{
  dim3 g((N + TN - 1) / TN, (M + TM - 1) / TM);
  gemm_wmma_bf16<<<g, dim3(256), 0, s>>>(A, W, bias, Cf, Cb, M, N, K,
                                         lda, ldw, ldc, arpb, abstr, abase,
                                         crpb, cbstr, cbase, wco, wtr, act, accum);
}

extern "C" void kernel_launch(void* const* d_in, const int* in_sizes, int n_in,
                              void* d_out, int out_size, void* d_ws, size_t ws_size,
                              hipStream_t stream)
{
  (void)in_sizes; (void)n_in; (void)out_size; (void)ws_size;

  const float* x        = (const float*)d_in[0];
  const float* patch_w  = (const float*)d_in[1];
  const float* patch_b  = (const float*)d_in[2];
  const float* cls_tok  = (const float*)d_in[3];
  const float* box_tok  = (const float*)d_in[4];
  const float* ln1_g    = (const float*)d_in[5];
  const float* ln1_b    = (const float*)d_in[6];
  const float* qkv_w    = (const float*)d_in[7];
  const float* tokqkv_w = (const float*)d_in[8];
  const float* proj_w   = (const float*)d_in[9];
  const float* proj_b   = (const float*)d_in[10];
  const float* tokproj_w= (const float*)d_in[11];
  const float* tokproj_b= (const float*)d_in[12];
  const float* ln2_g    = (const float*)d_in[13];
  const float* ln2_b    = (const float*)d_in[14];
  const float* ff1_w    = (const float*)d_in[15];
  const float* ff1_b    = (const float*)d_in[16];
  const float* ff2_w    = (const float*)d_in[17];
  const float* ff2_b    = (const float*)d_in[18];
  const float* lnf_g    = (const float*)d_in[19];
  const float* lnf_b    = (const float*)d_in[20];
  const float* clsh_w   = (const float*)d_in[21];
  const float* clsh_b   = (const float*)d_in[22];
  const float* bh1_w    = (const float*)d_in[23];
  const float* bh1_b    = (const float*)d_in[24];
  const float* bh2_w    = (const float*)d_in[25];
  const float* bh2_b    = (const float*)d_in[26];

  const int MP = BATCH * TOK;        // 3136 patch rows
  const int MA = BATCH * ROWS_IMG;   // 3168 all rows

  char* wsp = (char*)d_ws;
  size_t off = 0;
  auto carve = [&](size_t bytes) -> char* {
    char* p = wsp + off;
    off = (off + bytes + 255) & ~(size_t)255;
    return p;
  };
  float*  z      = (float*) carve((size_t)MA * E_DIM * 4);
  bf16_t* Apatch = (bf16_t*)carve((size_t)MP * 768 * 2);
  bf16_t* h_sp   = (bf16_t*)carve((size_t)BATCH * 2 * E_DIM * 2);
  bf16_t* h_pat  = (bf16_t*)carve((size_t)MP * E_DIM * 2);
  bf16_t* h_all  = (bf16_t*)carve((size_t)MA * E_DIM * 2);
  float*  qkvb   = (float*) carve((size_t)MP * 1152 * 4);
  bf16_t* attnb  = (bf16_t*)carve((size_t)MP * E_DIM * 2);
  float*  pof    = (float*) carve((size_t)MP * E_DIM * 4);
  bf16_t* pob    = (bf16_t*)carve((size_t)MP * E_DIM * 2);
  float*  tkvb   = (float*) carve((size_t)MP * 768 * 4);
  float*  tqb    = (float*) carve((size_t)BATCH * 2 * E_DIM * 4);
  bf16_t* sob    = (bf16_t*)carve((size_t)BATCH * 2 * E_DIM * 2);
  bf16_t* gbuf   = (bf16_t*)carve((size_t)MA * 1536 * 2);
  bf16_t* zfsp   = (bf16_t*)carve((size_t)BATCH * 2 * E_DIM * 2);
  bf16_t* ubuf   = (bf16_t*)carve((size_t)BATCH * 768 * 2);

  // --- embed ---
  init_tokens<<<BATCH, 128, 0, stream>>>(z, cls_tok, box_tok);
  im2col_patch<<<MP, 192, 0, stream>>>(x, Apatch);
  // z[b, 2+t, :] = patch(A) @ patch_w^T + patch_b
  gemm(stream, Apatch, patch_w, patch_b, z, nullptr,
       MP, E_DIM, 768, 768, 768, E_DIM,
       MP, 0, 0, TOK, ROWS_IMG, 2, 0, /*wtr=*/1, 0, 0);

  for (int l = 0; l < LAYERS; ++l) {
    const float* qw   = qkv_w    + (size_t)l * E_DIM * 1152;
    const float* tqw  = tokqkv_w + (size_t)l * E_DIM * 1152;
    const float* pw   = proj_w   + (size_t)l * E_DIM * E_DIM;
    const float* pb   = proj_b   + (size_t)l * E_DIM;
    const float* tpw  = tokproj_w+ (size_t)l * E_DIM * E_DIM;
    const float* tpb  = tokproj_b+ (size_t)l * E_DIM;
    const float* f1w  = ff1_w    + (size_t)l * E_DIM * 1536;
    const float* f1b  = ff1_b    + (size_t)l * 1536;
    const float* f2w  = ff2_w    + (size_t)l * 1536 * E_DIM;
    const float* f2b  = ff2_b    + (size_t)l * E_DIM;

    layernorm_k<<<MA, 128, 0, stream>>>(z, ln1_g + l * E_DIM, ln1_b + l * E_DIM,
                                        h_sp, h_pat, 1);
    // qkv = h_pat @ qkv_w[l]
    gemm(stream, h_pat, qw, nullptr, qkvb, nullptr,
         MP, 1152, E_DIM, E_DIM, 1152, 1152,
         MP, 0, 0, MP, 0, 0, 0, 0, 0, 0);
    local_attn<<<MP, dim3(32, NHEAD), 0, stream>>>(qkvb, attnb);
    // po = attn @ proj_w + proj_b  (f32 + bf16 copies)
    gemm(stream, attnb, pw, pb, pof, pob,
         MP, E_DIM, E_DIM, E_DIM, E_DIM, E_DIM,
         MP, 0, 0, MP, 0, 0, 0, 0, 0, 0);
    // tk,tv = po @ tokqkv_w[:, 384:1152]
    gemm(stream, pob, tqw, nullptr, tkvb, nullptr,
         MP, 768, E_DIM, E_DIM, 1152, 768,
         MP, 0, 0, MP, 0, 0, /*wco=*/384, 0, 0, 0);
    // tq = h_special @ tokqkv_w[:, 0:384]
    gemm(stream, h_sp, tqw, nullptr, tqb, nullptr,
         BATCH * 2, E_DIM, E_DIM, E_DIM, 1152, E_DIM,
         BATCH * 2, 0, 0, BATCH * 2, 0, 0, 0, 0, 0, 0);
    token_attn<<<BATCH * NHEAD, 256, 0, stream>>>(tqb, tkvb, sob);
    // z[b,{0,1},:] += so @ tokproj_w + tokproj_b
    gemm(stream, sob, tpw, tpb, z, nullptr,
         BATCH * 2, E_DIM, E_DIM, E_DIM, E_DIM, E_DIM,
         BATCH * 2, 0, 0, /*crpb=*/2, ROWS_IMG, 0, 0, 0, 0, /*accum=*/1);
    add_po<<<MP, 128, 0, stream>>>(z, pof);

    layernorm_k<<<MA, 128, 0, stream>>>(z, ln2_g + l * E_DIM, ln2_b + l * E_DIM,
                                        nullptr, h_all, 0);
    // g = gelu(h2 @ ff1_w + ff1_b)  (bf16 only)
    gemm(stream, h_all, f1w, f1b, nullptr, gbuf,
         MA, 1536, E_DIM, E_DIM, 1536, 1536,
         MA, 0, 0, MA, 0, 0, 0, 0, /*act=*/1, 0);
    // z += g @ ff2_w + ff2_b
    gemm(stream, gbuf, f2w, f2b, z, nullptr,
         MA, E_DIM, 1536, 1536, E_DIM, E_DIM,
         MA, 0, 0, MA, 0, 0, 0, 0, 0, /*accum=*/1);
  }

  // --- final LN + heads ---
  layernorm_k<<<MA, 128, 0, stream>>>(z, lnf_g, lnf_b, zfsp, nullptr, 1);
  // class_logits = zf[:,0] @ clsh_w + clsh_b  -> d_out[0 : 16*1000]
  gemm(stream, zfsp, clsh_w, clsh_b, (float*)d_out, nullptr,
       BATCH, 1000, E_DIM, E_DIM, 1000, 1000,
       /*arpb=*/1, 2, 0, BATCH, 0, 0, 0, 0, 0, 0);
  // u = gelu(zf[:,1] @ bh1_w + bh1_b)
  gemm(stream, zfsp, bh1_w, bh1_b, nullptr, ubuf,
       BATCH, 768, E_DIM, E_DIM, 768, 768,
       /*arpb=*/1, 2, 1, BATCH, 0, 0, 0, 0, /*act=*/1, 0);
  // box = sigmoid(u @ bh2_w + bh2_b) -> d_out[16000 : 16064]
  gemm(stream, ubuf, bh2_w, bh2_b, (float*)d_out + BATCH * 1000, nullptr,
       BATCH, 4, 768, 768, 4, 4,
       BATCH, 0, 0, BATCH, 0, 0, 0, 0, /*act=*/2, 0);
}
